// BertSelfAttention_13675175870904
// MI455X (gfx1250) — matrix operand order repaired and checked by reading, not compiled, not run on hardware
//
#include <hip/hip_runtime.h>
#include <hip/hip_bf16.h>

typedef __attribute__((ext_vector_type(16))) __bf16 v16bf;
typedef __attribute__((ext_vector_type(8)))  __bf16 v8bf;
typedef __attribute__((ext_vector_type(8)))  float  v8f;

#define S_LEN  1024
#define BATCH  4
#define HID    1024
#define NHEAD  16
#define HD     64
#define MROWS  (S_LEN*BATCH)   // 4096

// ---- helpers ---------------------------------------------------------------

static __device__ inline v16bf cat8(v8bf lo, v8bf hi) {
  v16bf r;
#pragma unroll
  for (int i = 0; i < 8; ++i) { r[i] = lo[i]; r[i + 8] = hi[i]; }
  return r;
}

// B fragment: 16 contiguous bf16 per lane (lane%16 = column, lane/16 selects K half)
static __device__ inline v16bf ld_bfrag(const __bf16* p) {
  return cat8(*(const v8bf*)p, *(const v8bf*)(p + 8));
}

// A fragment: 16 contiguous bf16 @p (K run {klo..klo+15})
static __device__ inline v16bf ld_afrag(const __bf16* p) {
  return cat8(*(const v8bf*)p, *(const v8bf*)(p + 8));
}

static __device__ inline v8f wmma_bf16(v16bf a, v16bf b, v8f c) {
  return __builtin_amdgcn_wmma_f32_16x16x32_bf16(false, a, false, b,
                                                 (short)0, c, false, false);
}

static __device__ inline v8bf pack8(float4 a, float4 b) {
  v8bf r;
  r[0] = (__bf16)a.x; r[1] = (__bf16)a.y; r[2] = (__bf16)a.z; r[3] = (__bf16)a.w;
  r[4] = (__bf16)b.x; r[5] = (__bf16)b.y; r[6] = (__bf16)b.z; r[7] = (__bf16)b.w;
  return r;
}

// async DMA: 16 bytes per lane, global -> LDS, tracked by ASYNCcnt
static __device__ inline void async_ld16(unsigned lds_byte_off, unsigned g_byte_off,
                                         unsigned long long g_base) {
  asm volatile("global_load_async_to_lds_b128 %0, %1, %2"
               :: "v"(lds_byte_off), "v"(g_byte_off), "s"(g_base) : "memory");
}

// ---- kernel 0: fp32 -> bf16 convert (8 elems / thread) ---------------------

__global__ __launch_bounds__(256)
void cvt_bf16(const float* __restrict__ src, __bf16* __restrict__ dst, int n8) {
  const int i = blockIdx.x * blockDim.x + threadIdx.x;
  if (i < n8) {
    const float4* s = (const float4*)src + (size_t)i * 2;
    float4 a = s[0], b = s[1];
    *((v8bf*)dst + i) = pack8(a, b);
  }
}

// ---- kernel 1: QKV projection (C = X * W^T + bias), bf16 WMMA --------------
// grid: (HID/128, MROWS/128)  block: 256 (8 waves)
// MODE 0 -> Qb (S,B,H) ; MODE 1 -> Kb (S,B,H) ; MODE 2 -> Vt[head][d][t]
// LDS tiles filled with GLOBAL_LOAD_ASYNC_TO_LDS_B128, double buffered.

#define LDS_STRIDE 56                       // bf16/row: 112B = 28-bank stride
#define BUF_BYTES  (128 * LDS_STRIDE * 2)   // 14336

template <int MODE>
__global__ __launch_bounds__(256)
void qkv_gemm(const __bf16* __restrict__ Xb, const __bf16* __restrict__ Wb,
              const float* __restrict__ bias, __bf16* __restrict__ Out)
{
  __shared__ __bf16 As[2][128 * LDS_STRIDE];
  __shared__ __bf16 Bs[2][128 * LDS_STRIDE];

  const int n0 = blockIdx.x * 128;
  const int m0 = blockIdx.y * 128;
  const int tid  = threadIdx.x;
  const int lane = tid & 31;
  const int wv   = tid >> 5;
  const int wm   = wv & 3;   // 4 wave-rows (32 M each)
  const int wn   = wv >> 2;  // 2 wave-cols (64 N each)

  const int l15 = lane & 15;
  const int klo = (lane < 16) ? 0 : 16;
  const int kb  = (lane < 16) ? 0 : 16;

  // async-fill decomposition: each thread owns one 16B chunk of two rows
  const unsigned c  = tid & 3;    // chunk within 64B row (32 bf16)
  const unsigned rr = tid >> 2;   // rows 0..63 (+64 for second half)
  const unsigned as_base = (unsigned)(size_t)(void*)&As[0][0];
  const unsigned bs_base = (unsigned)(size_t)(void*)&Bs[0][0];
  const unsigned long long xbase = (unsigned long long)Xb;
  const unsigned long long wbase = (unsigned long long)Wb;

  auto issue = [&](int bufi, int k0) {
    const unsigned al = as_base + (unsigned)bufi * BUF_BYTES + rr * 112u + c * 16u;
    const unsigned ag = ((unsigned)((m0 + (int)rr) * HID + k0) + c * 8u) * 2u;
    async_ld16(al,              ag,                  xbase);
    async_ld16(al + 64u * 112u, ag + 64u * HID * 2u, xbase);
    const unsigned bl = bs_base + (unsigned)bufi * BUF_BYTES + rr * 112u + c * 16u;
    const unsigned bg = ((unsigned)((n0 + (int)rr) * HID + k0) + c * 8u) * 2u;
    async_ld16(bl,              bg,                  wbase);
    async_ld16(bl + 64u * 112u, bg + 64u * HID * 2u, wbase);
  };

  v8f acc[2][4];
#pragma unroll
  for (int i = 0; i < 2; ++i)
#pragma unroll
    for (int j = 0; j < 4; ++j) acc[i][j] = (v8f){0.f,0.f,0.f,0.f,0.f,0.f,0.f,0.f};

  // prologue: fill buffer 0
  issue(0, 0);
  asm volatile("s_wait_asynccnt 0" ::: "memory");
  __syncthreads();

  int buf = 0;
  for (int k0 = 0; k0 < HID; k0 += 32) {
    if (k0 + 32 < HID) issue(buf ^ 1, k0 + 32);   // prefetch next tile (DMA)

    v16bf af[2], bfm[4];
#pragma unroll
    for (int i = 0; i < 2; ++i)
      af[i] = ld_afrag(&As[buf][(wm * 32 + i * 16 + l15) * LDS_STRIDE + klo]);
#pragma unroll
    for (int j = 0; j < 4; ++j)
      bfm[j] = ld_bfrag(&Bs[buf][(wn * 64 + j * 16 + l15) * LDS_STRIDE + kb]);

#pragma unroll
    for (int i = 0; i < 2; ++i)
#pragma unroll
      for (int j = 0; j < 4; ++j)
        acc[i][j] = wmma_bf16(af[i], bfm[j], acc[i][j]);

    __syncthreads();                              // everyone done reading buf
    if (k0 + 32 < HID) {
      asm volatile("s_wait_asynccnt 0" ::: "memory");  // own DMA done
      __syncthreads();                                  // all waves' DMA visible
    }
    buf ^= 1;
  }

  // epilogue: bias + store (bf16); MODE resolved at compile time
  const int hi8 = (lane >> 4) << 3;
#pragma unroll
  for (int i = 0; i < 2; ++i) {
#pragma unroll
    for (int j = 0; j < 4; ++j) {
      const int n = n0 + wn * 64 + j * 16 + l15;
      const float bval = bias[n];
#pragma unroll
      for (int r = 0; r < 8; ++r) {
        const int m = m0 + wm * 32 + i * 16 + r + hi8;
        const float v = acc[i][j][r] + bval;
        if (MODE < 2) {
          Out[(size_t)m * HID + n] = (__bf16)v;
        } else {
          // V transposed per head: Vt[(head*64 + d)*S + t]
          const int bb = m & 3, ss = m >> 2;       // m = s*B + b
          const int nh = (bb << 4) | (n >> 6);     // head = b*16 + h
          const int dd = n & 63;
          Out[((size_t)(nh * HD + dd)) * S_LEN + ss] = (__bf16)v;
        }
      }
    }
  }
}

// ---- kernel 2: flash attention per head ------------------------------------
// grid: (S/64, B*NHEAD)  block: 128 (4 waves, 16 query rows each)

__global__ __launch_bounds__(128)
void attn_kernel(const __bf16* __restrict__ Qb, const __bf16* __restrict__ Kb,
                 const __bf16* __restrict__ Vt, const float* __restrict__ mask,
                 float* __restrict__ out)
{
  __shared__ __bf16 Pl[4 * 16 * 88];   // per-wave 16x64 P tile, stride 88 (conflict-free)

  const int n  = blockIdx.y;           // head index = b*16 + h
  const int q0 = blockIdx.x * 64;
  const int b  = n >> 4, h = n & 15;
  const int lane = threadIdx.x & 31, w = threadIdx.x >> 5;
  const int l15  = lane & 15;
  const int klo  = (lane < 16) ? 0 : 16;
  const int kb16 = (lane < 16) ? 0 : 16;
  const float scale = 0.125f;          // 1/sqrt(64)

  const __bf16* qp = Qb + ((size_t)(q0 + w * 16 + l15) * BATCH + b) * HID + h * HD;
  const v16bf qlo = ld_afrag(qp + klo);
  const v16bf qhi = ld_afrag(qp + 32 + klo);

  v8f  o[4];
  float mi[8], li[8];
#pragma unroll
  for (int j = 0; j < 4; ++j) o[j] = (v8f){0.f,0.f,0.f,0.f,0.f,0.f,0.f,0.f};
#pragma unroll
  for (int r = 0; r < 8; ++r) { mi[r] = -1e30f; li[r] = 0.f; }

  __bf16* pw = &Pl[w * 16 * 88];

  for (int kt = 0; kt < S_LEN; kt += 64) {
    // ---- scores S = Q K^T * scale + mask -----------------------------------
    v8f s[4];
#pragma unroll
    for (int j = 0; j < 4; ++j) {
      const int key = kt + j * 16 + l15;
      const __bf16* kp = Kb + ((size_t)key * BATCH + b) * HID + h * HD;
      v16bf k0 = ld_bfrag(kp + kb16);
      v16bf k1 = ld_bfrag(kp + 32 + kb16);
      v8f cc = (v8f){0.f,0.f,0.f,0.f,0.f,0.f,0.f,0.f};
      cc = wmma_bf16(qlo, k0, cc);
      cc = wmma_bf16(qhi, k1, cc);
      const float mk = mask[(size_t)n * S_LEN + kt + j * 16 + l15];
#pragma unroll
      for (int r = 0; r < 8; ++r) cc[r] = cc[r] * scale + mk;
      s[j] = cc;
    }

    // ---- online softmax (per C-fragment row) -------------------------------
#pragma unroll
    for (int r = 0; r < 8; ++r) {
      float v = fmaxf(fmaxf(s[0][r], s[1][r]), fmaxf(s[2][r], s[3][r]));
      v = fmaxf(v, __shfl_xor(v, 1, 16));
      v = fmaxf(v, __shfl_xor(v, 2, 16));
      v = fmaxf(v, __shfl_xor(v, 4, 16));
      v = fmaxf(v, __shfl_xor(v, 8, 16));
      const float mn   = fmaxf(mi[r], v);
      const float corr = __expf(mi[r] - mn);
      mi[r] = mn;
      li[r] *= corr;
#pragma unroll
      for (int j = 0; j < 4; ++j) o[j][r] *= corr;
      float rs = 0.f;
#pragma unroll
      for (int j = 0; j < 4; ++j) {
        const float p = __expf(s[j][r] - mn);
        s[j][r] = p;
        rs += p;
      }
      rs += __shfl_xor(rs, 1, 16);
      rs += __shfl_xor(rs, 2, 16);
      rs += __shfl_xor(rs, 4, 16);
      rs += __shfl_xor(rs, 8, 16);
      li[r] += rs;
    }

    // ---- re-shape P: C-layout -> A-fragments via per-wave LDS --------------
    const int hi8 = (lane >> 4) << 3;
#pragma unroll
    for (int j = 0; j < 4; ++j)
#pragma unroll
      for (int r = 0; r < 8; ++r)
        pw[(r + hi8) * 88 + j * 16 + l15] = (__bf16)s[j][r];

    asm volatile("s_wait_dscnt 0" ::: "memory");  // wave-private RAW through LDS

    const v16bf p0 = ld_afrag(&pw[l15 * 88 + klo]);
    const v16bf p1 = ld_afrag(&pw[l15 * 88 + 32 + klo]);

    // ---- O += P V ----------------------------------------------------------
#pragma unroll
    for (int j = 0; j < 4; ++j) {
      const __bf16* vp = Vt + ((size_t)(n * HD + j * 16 + l15)) * S_LEN + kt;
      v16bf v0 = ld_bfrag(vp + kb16);
      v16bf v1 = ld_bfrag(vp + 32 + kb16);
      o[j] = wmma_bf16(p0, v0, o[j]);
      o[j] = wmma_bf16(p1, v1, o[j]);
    }
  }

  // ---- normalize and scatter to (S, B, H) fp32 -----------------------------
#pragma unroll
  for (int r = 0; r < 8; ++r) li[r] = 1.f / li[r];
  const int hi8 = (lane >> 4) << 3;
#pragma unroll
  for (int j = 0; j < 4; ++j)
#pragma unroll
    for (int r = 0; r < 8; ++r) {
      const int srow = q0 + w * 16 + r + hi8;
      out[((size_t)srow * BATCH + b) * HID + h * HD + j * 16 + l15] = o[j][r] * li[r];
    }
}

// ---- host launcher ---------------------------------------------------------

extern "C" void kernel_launch(void* const* d_in, const int* in_sizes, int n_in,
                              void* d_out, int out_size, void* d_ws, size_t ws_size,
                              hipStream_t stream)
{
  const float* hidden = (const float*)d_in[0];
  const float* mask   = (const float*)d_in[1];
  const float* Wq     = (const float*)d_in[2];
  const float* bq     = (const float*)d_in[3];
  const float* Wk     = (const float*)d_in[4];
  const float* bk     = (const float*)d_in[5];
  const float* Wv     = (const float*)d_in[6];
  const float* bv     = (const float*)d_in[7];

  __bf16* Qb = (__bf16*)d_ws;                         // 4096x1024 bf16 = 8MB
  __bf16* Kb = Qb + (size_t)MROWS * HID;              // 8MB
  __bf16* Vt = Kb + (size_t)MROWS * HID;              // 8MB (per-head transposed V)
  __bf16* Xb = Vt + (size_t)MROWS * HID;              // 8MB  bf16 copy of X
  __bf16* Wb = Xb + (size_t)MROWS * HID;              // 3 x 2MB bf16 weights

  // fp32 -> bf16 copies (inputs stay untouched)
  cvt_bf16<<<(MROWS * HID / 8 + 255) / 256, 256, 0, stream>>>(hidden, Xb, MROWS * HID / 8);
  cvt_bf16<<<(HID * HID / 8 + 255) / 256, 256, 0, stream>>>(Wq, Wb + 0 * (size_t)HID * HID, HID * HID / 8);
  cvt_bf16<<<(HID * HID / 8 + 255) / 256, 256, 0, stream>>>(Wk, Wb + 1 * (size_t)HID * HID, HID * HID / 8);
  cvt_bf16<<<(HID * HID / 8 + 255) / 256, 256, 0, stream>>>(Wv, Wb + 2 * (size_t)HID * HID, HID * HID / 8);

  dim3 ggrid(HID / 128, MROWS / 128);
  qkv_gemm<0><<<ggrid, 256, 0, stream>>>(Xb, Wb + 0 * (size_t)HID * HID, bq, Qb);
  qkv_gemm<1><<<ggrid, 256, 0, stream>>>(Xb, Wb + 1 * (size_t)HID * HID, bk, Kb);
  qkv_gemm<2><<<ggrid, 256, 0, stream>>>(Xb, Wb + 2 * (size_t)HID * HID, bv, Vt);

  attn_kernel<<<dim3(S_LEN / 64, BATCH * NHEAD), 128, 0, stream>>>(
      Qb, Kb, Vt, mask, (float*)d_out);
}
